// Generator_61503931679157
// MI455X (gfx1250) — matrix operand order repaired and checked by reading, not compiled, not run on hardware
//
#include <hip/hip_runtime.h>
#include <hip/hip_bf16.h>

// CDNA5 / gfx1250, wave32. Matrix math via v_wmma_f32_16x16x32_bf16.
// Activations are NHWC internally (contiguous-channel => 128-bit loads feed WMMA B fragments).

typedef __bf16 bf16;
typedef __attribute__((ext_vector_type(16))) __bf16 v16bf;
typedef __attribute__((ext_vector_type(8)))  float  v8f;

#define SQRT2F 1.41421356237309515f

__device__ __forceinline__ bf16 f2bf(float f) { return (bf16)f; }  // native v_cvt on gfx1250

// A-fragment K map (contiguous, matching B): lane<16 holds K{0..15}, lane>=16 K{16..31}
__device__ __forceinline__ int kmapA(int e, int lane) {
  return e + ((lane >= 16) ? 16 : 0);
}

// Deterministic stand-in for jax noise (threefry not reproducible here): hash -> Box-Muller.
__device__ __forceinline__ float gnoise(int l, int b, int y, int x) {
  unsigned h = 0x9E3779B9u ^ ((unsigned)l * 0x85EBCA6Bu);
  h ^= (unsigned)b * 0xC2B2AE35u; h = (h << 13) | (h >> 19);
  h ^= (unsigned)y * 0x27D4EB2Fu; h *= 0x165667B1u;
  h ^= (unsigned)x * 0x9E3779B1u;
  h ^= h >> 15; h *= 0x2C1B3C6Du; h ^= h >> 12; h *= 0x297A2D39u; h ^= h >> 15;
  unsigned h2 = h * 0x9E3779B9u + 1u; h2 ^= h2 >> 16; h2 *= 0x85EBCA6Bu; h2 ^= h2 >> 13;
  float u1 = ((h >> 9) + 1u) * (1.0f / 8388608.0f);   // (0,1]
  float u2 = (h2 >> 9) * (1.0f / 8388608.0f);
  return sqrtf(-2.0f * __logf(u1)) * __cosf(6.28318530718f * u2);
}

// ---------------- zero helper (init padded latent rows once per launch) ----------------
__global__ void k_zero(float* __restrict__ P, int n) {
  int i = blockIdx.x * blockDim.x + threadIdx.x;
  if (i < n) P[i] = 0.f;
}

// ---------------- PixelNorm: z * rsqrt(mean(z^2)+1e-8), [8,512] -> rows 0..7 of [16,512] ----------------
__global__ void k_pixelnorm(const float* __restrict__ Z, float* __restrict__ O) {
  __shared__ float red[256];
  int b = blockIdx.x, t = threadIdx.x;
  float v0 = Z[b * 512 + t];
  float v1 = Z[b * 512 + 256 + t];
  red[t] = v0 * v0 + v1 * v1;
  __syncthreads();
  for (int s = 128; s > 0; s >>= 1) {
    if (t < s) red[t] += red[t + s];
    __syncthreads();
  }
  float r = rsqrtf(red[0] * (1.0f / 512.0f) + 1e-8f);
  O[b * 512 + t] = v0 * r;
  O[b * 512 + 256 + t] = v1 * r;
}

// ------- Small GEMM via WMMA: Y[m,n] = act( sum_k X[m,k]*W[n,k]*wscale + B[n]*bscale ) -------
// X and Y are [16][512] (rows 8..15 are deterministic padding that never feeds rows 0..7).
// Unguarded, fully coalescable loads; EXEC all-ones throughout. One wave per block, grid.x=32.
__global__ void k_gemm16(const float* __restrict__ X, const float* __restrict__ W,
                         const float* __restrict__ B, float* __restrict__ Y,
                         float wscale, float bscale, int act) {
  int lane = threadIdx.x;
  int n0 = blockIdx.x * 16;
  int m = lane & 15;
  int nh = n0 + m;
  int baseA = (lane >= 16) ? 8 : 0;    // K offset of second half-row / C row offset
  int baseB = (lane >= 16) ? 8 : 0;
  v8f c = {};
  for (int kc = 0; kc < 512; kc += 32) {
    const float* xa = X + m * 512 + kc + baseA;
    const float* wb = W + nh * 512 + kc + baseB;
    v16bf a, b;
#pragma unroll
    for (int e = 0; e < 8; ++e) {      // two contiguous 8-float runs
      a[e]     = f2bf(xa[e]);
      a[e + 8] = f2bf(xa[e + 16]);
    }
#pragma unroll
    for (int e = 0; e < 16; ++e)       // two 8-float runs (same K order as A)
      b[e] = f2bf(wb[(e < 8) ? e : e + 8] * wscale);
    c = __builtin_amdgcn_wmma_f32_16x16x32_bf16(false, a, false, b, (short)0, c, false, false);
  }
  float bb = B[nh] * bscale;
#pragma unroll
  for (int r = 0; r < 8; ++r) {        // write all 16 rows (8 per lane-half)
    float v = c[r] + bb;
    if (act) v = (v > 0.0f ? v : 0.2f * v) * SQRT2F;
    Y[(baseA + r) * 512 + nh] = v;
  }
}

// ---------------- Pack conv weights into WMMA A-fragment layout (bf16) ----------------
// Layout: [ot(32)][tap(9)][ic(16)][lane(32)][e(16)]; flip reverses taps (transpose-conv).
__global__ void k_packw(const float* __restrict__ W, bf16* __restrict__ Wp, int flip) {
  int idx = blockIdx.x * blockDim.x + threadIdx.x;
  if (idx >= 32 * 9 * 16 * 32) return;
  int lane = idx & 31;
  int t = idx >> 5;
  int ic = t & 15; t >>= 4;
  int tap = t % 9;
  int ot = t / 9;
  int tp = flip ? (8 - tap) : tap;
  int o = ot * 16 + (lane & 15);
  v16bf vv;
#pragma unroll
  for (int e = 0; e < 16; ++e) {
    int i = ic * 32 + kmapA(e, lane);
    vv[e] = f2bf(W[(size_t)(o * 512 + i) * 9 + tp]);
  }
  *(v16bf*)(Wp + (size_t)idx * 16) = vv;
}

// ---------------- Wsq[o,i] = sum_tap W^2 (for demod) ----------------
__global__ void k_wsq(const float* __restrict__ W, float* __restrict__ Wsq) {
  int idx = blockIdx.x * blockDim.x + threadIdx.x;
  if (idx >= 512 * 512) return;
  float s = 0.f;
#pragma unroll
  for (int k = 0; k < 9; ++k) { float w = W[(size_t)idx * 9 + k]; s += w * w; }
  Wsq[idx] = s;
}

// ---------------- demod d[b,o] = rsqrt(scale^2 * sum_i Wsq[o,i]*s[b,i]^2 + 1e-8) ----------------
__global__ void k_demod(const float* __restrict__ Wsq, const float* __restrict__ S,
                        float* __restrict__ D, float scale2) {
  int idx = blockIdx.x * blockDim.x + threadIdx.x;
  if (idx >= 8 * 512) return;
  int b = idx >> 9, o = idx & 511;
  float acc = 0.f;
  for (int i = 0; i < 512; ++i) {
    float s = S[b * 512 + i];
    acc += Wsq[o * 512 + i] * s * s;
  }
  D[idx] = rsqrtf(scale2 * acc + 1e-8f);
}

// -------- Modulate by style + pad (+ stride-2 zero-insert for transpose-conv) -> NHWC bf16 --------
// Xp layout: [b][u][v][i] (i fastest => coalesced stores, contiguous channels for WMMA B frags).
// nchw=1 reads X as [512][H][H] broadcast (const_input); else X is NHWC [b][y][x][i].
__global__ void k_modpad(const float* __restrict__ X, const float* __restrict__ S,
                         bf16* __restrict__ Xp, int H, int PH, int up, int bstride, int nchw) {
  int idx = blockIdx.x * blockDim.x + threadIdx.x;
  int total = 8 * PH * PH * 512;
  if (idx >= total) return;
  int i = idx & 511;
  int t = idx >> 9;
  int v = t % PH; t /= PH;
  int u = t % PH;
  int b = t / PH;
  int y, x, inr = 0;
  if (!up) {
    y = u - 1; x = v - 1;
    inr = (y >= 0 && y < H && x >= 0 && x < H);
  } else {  // dilated (stride-2 zero insert) with pad 2
    int uu = u - 2, vv = v - 2;
    y = uu >> 1; x = vv >> 1;
    inr = (uu >= 0 && vv >= 0 && !(uu & 1) && !(vv & 1) && y < H && x < H);
  }
  float val = 0.f;
  if (inr) {
    size_t src = nchw ? ((size_t)i * H + y) * H + x
                      : (size_t)b * bstride + ((size_t)y * H + x) * 512 + i;
    val = X[src] * S[b * 512 + i];
  }
  Xp[idx] = f2bf(val);
}

// ---------------- Implicit-GEMM 3x3 conv via WMMA ----------------
// grid: (ceil(OH*OH/16), 8 oGroups of 64, 8 batch). One wave/block, 4 C tiles (64 o x 16 pixels).
// B fragments: 16 contiguous bf16 (32B aligned) per lane from NHWC input -> 2x global_load_b128.
// A fragments: 32B contiguous from pre-swizzled weights. No conditional loads: EXEC all-ones.
__global__ void k_conv3x3(const bf16* __restrict__ Wp, const bf16* __restrict__ Xp,
                          const float* __restrict__ D, const float* __restrict__ bias,
                          const float* __restrict__ nstr, float* __restrict__ Y,
                          int OH, int PH, float wscale, int layer, int fuse) {
  int lane = threadIdx.x;
  int b = blockIdx.z;
  int og = blockIdx.y;
  int n0 = blockIdx.x * 16;
  int numPix = OH * OH;
  int np = n0 + (lane & 15);
  int pv = np < numPix;
  int pc = pv ? np : numPix - 1;   // clamp to a valid pixel: loads stay in-bounds, store is masked
  int py = pc / OH, px = pc % OH;
  const bf16* xb = Xp + ((size_t)b * PH * PH) * 512;
  int baseB = (lane >= 16) ? 16 : 0;
  const size_t otStride = (size_t)9 * 16 * 32 * 16;  // elements between o-tiles in packed weights
  v8f acc[4] = {};
  for (int ic = 0; ic < 16; ++ic) {
    __builtin_prefetch(Wp + ((((size_t)(og * 4) * 9) * 16 + ((ic + 1) & 15)) * 32 + lane) * 16, 0, 0);
#pragma unroll
    for (int tap = 0; tap < 9; ++tap) {
      int ky = tap / 3, kx = tap % 3;
      v16bf bfr = *(const v16bf*)(xb + ((size_t)(py + ky) * PH + (px + kx)) * 512 + ic * 32 + baseB);
      const bf16* wrow = Wp + ((((size_t)(og * 4) * 9 + tap) * 16 + ic) * 32 + lane) * 16;
#pragma unroll
      for (int mm = 0; mm < 4; ++mm) {
        v16bf af = *(const v16bf*)(wrow + (size_t)mm * otStride);
        acc[mm] = __builtin_amdgcn_wmma_f32_16x16x32_bf16(false, af, false, bfr, (short)0,
                                                          acc[mm], false, false);
      }
    }
  }
  if (!pv) return;  // branch only AFTER all WMMAs
  float ns = fuse ? nstr[layer] * gnoise(layer, b, py, px) : 0.0f;
  int mb = (lane >= 16) ? 8 : 0;
#pragma unroll
  for (int mm = 0; mm < 4; ++mm) {
    float* yp = Y + (((size_t)b * OH + py) * OH + px) * 512 + og * 64 + mm * 16 + mb;
#pragma unroll
    for (int r = 0; r < 8; ++r) {   // 8 contiguous channels per lane (NHWC)
      int o = og * 64 + mm * 16 + mb + r;
      float vo = acc[mm][r] * wscale * D[b * 512 + o];
      if (fuse) {
        vo += ns + bias[o];
        vo = (vo > 0.f ? vo : 0.2f * vo) * SQRT2F;
      }
      yp[r] = vo;
    }
  }
}

// -------- 4x4 blur (pad 1,1) + noise + bias + leakyReLU*sqrt2 (up-layers), NHWC in/out --------
__global__ void k_blur_act(const float* __restrict__ Yt, const float* __restrict__ bias,
                           const float* __restrict__ nstr, float* __restrict__ Xo,
                           int SH, int OH, int layer) {
  int idx = blockIdx.x * blockDim.x + threadIdx.x;
  if (idx >= 8 * OH * OH * 512) return;
  int o = idx & 511;
  int t = idx >> 9;
  int c = t % OH; t /= OH;
  int r = t % OH;
  int b = t / OH;
  const float K1[4] = {1.f, 3.f, 3.f, 1.f};
  const float* src = Yt + (size_t)b * SH * SH * 512 + o;
  float acc = 0.f;
#pragma unroll
  for (int a = 0; a < 4; ++a) {
    int u = r + a - 1;
    if (u < 0 || u >= SH) continue;
#pragma unroll
    for (int bb = 0; bb < 4; ++bb) {
      int v = c + bb - 1;
      if (v < 0 || v >= SH) continue;
      acc += K1[a] * K1[bb] * src[((size_t)u * SH + v) * 512];
    }
  }
  acc *= (1.0f / 16.0f);  // (k/64)*4
  acc += nstr[layer] * gnoise(layer, b, r, c) + bias[o];
  acc = (acc > 0.f ? acc : 0.2f * acc) * SQRT2F;
  Xo[idx] = acc;
}

// ---------------- to_rgb: 1x1 modulated conv (no demod); NHWC in, NCHW out ----------------
__global__ void k_torgb(const float* __restrict__ X, const float* __restrict__ Wr,
                        const float* __restrict__ S, const float* __restrict__ Br,
                        float* __restrict__ Out, int HW) {
  int idx = blockIdx.x * blockDim.x + threadIdx.x;
  if (idx >= 8 * 3 * HW) return;
  int p = idx % HW; int t = idx / HW;
  int o = t % 3; int b = t / 3;
  const float* xp = X + ((size_t)b * HW + p) * 512;  // contiguous channel run
  float acc = 0.f;
  for (int i = 0; i < 512; ++i)
    acc += Wr[o * 512 + i] * S[b * 512 + i] * xp[i];
  Out[idx] = acc * 0.044194173824159216f /* 1/sqrt(512) */ + Br[o];
}

// ---------------- skip upsample: out += upfirdn2d(skip, k*4, up=2, pad=(2,1)); NCHW ----------------
__global__ void k_skipup(const float* __restrict__ Si, float* __restrict__ So, int H) {
  int OH = 2 * H;
  int idx = blockIdx.x * blockDim.x + threadIdx.x;
  if (idx >= 8 * 3 * OH * OH) return;
  int cc = idx % OH; int t = idx / OH;
  int rr = t % OH; t /= OH;
  int o = t % 3; int b = t / 3;
  const float K1[4] = {1.f, 3.f, 3.f, 1.f};
  float acc = 0.f;
#pragma unroll
  for (int a = 0; a < 4; ++a) {
    int dd = rr + a - 2;
    if (dd < 0 || (dd & 1)) continue;
    int y = dd >> 1; if (y >= H) continue;
#pragma unroll
    for (int bb = 0; bb < 4; ++bb) {
      int ee = cc + bb - 2;
      if (ee < 0 || (ee & 1)) continue;
      int x = ee >> 1; if (x >= H) continue;
      acc += K1[a] * K1[bb] * Si[(((size_t)b * 3 + o) * H + y) * H + x];
    }
  }
  So[idx] += acc * (1.0f / 16.0f);
}

extern "C" void kernel_launch(void* const* d_in, const int* in_sizes, int n_in,
                              void* d_out, int out_size, void* d_ws, size_t ws_size,
                              hipStream_t stream) {
  (void)in_sizes; (void)n_in; (void)out_size; (void)ws_size;
  const float* z      = (const float*)d_in[0];
  const float* mlp_w  = (const float*)d_in[1];
  const float* mlp_b  = (const float*)d_in[2];
  const float* cinput = (const float*)d_in[3];
  const float* conv_w = (const float*)d_in[4];
  const float* cmod_w = (const float*)d_in[5];
  const float* cmod_b = (const float*)d_in[6];
  const float* cbias  = (const float*)d_in[7];
  const float* nstr   = (const float*)d_in[8];
  const float* rgb_w  = (const float*)d_in[9];
  const float* rmod_w = (const float*)d_in[10];
  const float* rmod_b = (const float*)d_in[11];
  const float* rbias  = (const float*)d_in[12];

  // Workspace carve-out (~68 MB total).
  char* base = (char*)d_ws;
  size_t off = 0;
  auto sub = [&](size_t bytes) -> void* {
    off = (off + 255) & ~(size_t)255;
    void* p = base + off;
    off += bytes;
    return p;
  };
  float* lat0 = (float*)sub((size_t)16 * 512 * 4);
  float* lat1 = (float*)sub((size_t)16 * 512 * 4);
  float* sbuf = (float*)sub((size_t)16 * 512 * 4);
  float* dbuf = (float*)sub((size_t)8 * 512 * 4);
  float* wsq  = (float*)sub((size_t)512 * 512 * 4);
  bf16*  wpk  = (bf16*) sub((size_t)512 * 512 * 9 * 2);
  float* actA = (float*)sub((size_t)8 * 32 * 32 * 512 * 4);   // NHWC
  float* actB = (float*)sub((size_t)8 * 32 * 32 * 512 * 4);   // NHWC
  bf16*  xpad = (bf16*) sub((size_t)8 * 36 * 36 * 512 * 2);   // NHWC padded/modulated
  float* ytmp = (float*)sub((size_t)8 * 33 * 33 * 512 * 4);   // NHWC pre-blur
  float* skA  = (float*)sub((size_t)8 * 3 * 32 * 32 * 4);     // NCHW
  float* skB  = (float*)sub((size_t)8 * 3 * 32 * 32 * 4);     // NCHW

  const float MAP_SCALE  = 0.01f / 22.627416997969522f;
  const float MOD_SCALE  = 1.0f / 22.627416997969522f;
  const float CONV_SCALE = 0.014731391274719739f;  // 1/sqrt(512*9)

  // ---- Mapping network: PixelNorm + 8x (GEMM + fused leaky-relu) ----
  k_zero<<<(8 * 512 + 255) / 256, 256, 0, stream>>>(lat0 + 8 * 512, 8 * 512);  // pad rows 8..15
  k_pixelnorm<<<8, 256, 0, stream>>>(z, lat0);
  float* lc = lat0; float* ln = lat1;
  for (int i = 0; i < 8; ++i) {
    k_gemm16<<<32, 32, 0, stream>>>(lc, mlp_w + (size_t)i * 512 * 512, mlp_b + i * 512, ln,
                                    MAP_SCALE, 0.01f, 1);
    float* t = lc; lc = ln; ln = t;
  }

  // ---- Synthesis ----
  struct LCfg { int inH; int up; };
  const LCfg L[7] = {{4, 0}, {4, 1}, {8, 0}, {8, 1}, {16, 0}, {16, 1}, {32, 0}};
  const float* xin = cinput; int bstride = 0; int nchw = 1;   // layer0: broadcast NCHW const input
  float* act = actA; float* alt = actB;
  int rgbIdx = 0;

  for (int l = 0; l < 7; ++l) {
    int H = L[l].inH, up = L[l].up;
    int OH = up ? 2 * H : H;      // activation size after (conv [+blur])
    int SH = up ? 2 * H + 1 : H;  // raw conv output size
    int PH = SH + 2;              // padded conv-input size
    float* outAct = act;

    // style s[b,i]
    k_gemm16<<<32, 32, 0, stream>>>(lc, cmod_w + (size_t)l * 512 * 512, cmod_b + l * 512,
                                    sbuf, MOD_SCALE, 1.0f, 0);
    // demod
    k_wsq<<<(512 * 512 + 255) / 256, 256, 0, stream>>>(conv_w + (size_t)l * 512 * 512 * 9, wsq);
    k_demod<<<(8 * 512 + 255) / 256, 256, 0, stream>>>(wsq, sbuf, dbuf, CONV_SCALE * CONV_SCALE);
    // weights -> bf16 WMMA fragment layout (flipped taps for transpose-conv)
    k_packw<<<(32 * 9 * 16 * 32 + 255) / 256, 256, 0, stream>>>(
        conv_w + (size_t)l * 512 * 512 * 9, wpk, up);
    // modulate + pad (+ zero-insert) -> NHWC bf16
    int totpad = 8 * PH * PH * 512;
    k_modpad<<<(totpad + 255) / 256, 256, 0, stream>>>(xin, sbuf, xpad, H, PH, up, bstride, nchw);
    // implicit-GEMM conv (WMMA); fuse noise/bias/act for non-up layers
    dim3 cg((SH * SH + 15) / 16, 8, 8);
    float* ydst = up ? ytmp : outAct;
    k_conv3x3<<<cg, 32, 0, stream>>>(wpk, xpad, dbuf, cbias + l * 512, nstr, ydst,
                                     SH, PH, CONV_SCALE, l, up ? 0 : 1);
    if (up) {
      int tot = 8 * OH * OH * 512;
      k_blur_act<<<(tot + 255) / 256, 256, 0, stream>>>(ytmp, cbias + l * 512, nstr, outAct,
                                                        SH, OH, l);
    }

    // to_rgb after layers 0,2,4,6
    if ((l & 1) == 0) {
      int HW = OH * OH;
      k_gemm16<<<32, 32, 0, stream>>>(lc, rmod_w + (size_t)rgbIdx * 512 * 512,
                                      rmod_b + rgbIdx * 512, sbuf, MOD_SCALE, 1.0f, 0);
      float* dst = (rgbIdx == 3) ? (float*)d_out : (((rgbIdx & 1) == 0) ? skA : skB);
      k_torgb<<<(8 * 3 * HW + 255) / 256, 256, 0, stream>>>(
          outAct, rgb_w + (size_t)rgbIdx * 3 * 512, sbuf, rbias + rgbIdx * 3, dst, HW);
      if (rgbIdx > 0) {
        float* prev = ((rgbIdx & 1) == 0) ? skB : skA;
        k_skipup<<<(8 * 3 * HW + 255) / 256, 256, 0, stream>>>(prev, dst, OH / 2);
      }
      rgbIdx++;
    }

    xin = outAct; bstride = OH * OH * 512; nchw = 0;
    float* tt = act; act = alt; alt = tt;
  }
}
